// CondConvResidual_48241072668980
// MI455X (gfx1250) — matrix operand order repaired and checked by reading, not compiled, not run on hardware
//
#include <hip/hip_runtime.h>
#include <stdint.h>

// ---------------------------------------------------------------------------
// CondConv residual block for MI455X (gfx1250, wave32, WMMA).
// v2: removed all intra-GEMM barriers by preloading full K (pw1) / full B (pw2)
// into LDS once per block. Inner loops are pure ds_load_b128 + v_wmma.
// Workspace requirement: ~82.4 MB.
// ---------------------------------------------------------------------------

typedef __attribute__((ext_vector_type(16))) __bf16 v16bf;
typedef __attribute__((ext_vector_type(8)))  __bf16 v8bf;
typedef __attribute__((ext_vector_type(8)))  float  v8f;

#define EPS_ 1e-3f
#define B_  32
#define HH_ 56
#define WW_ 56
#define C_  96
#define E_  4
#define M_  384
#define F_  96
#define P_  (HH_*WW_)   // 3136

__device__ __forceinline__ __bf16 f2bf(float x){ return (__bf16)x; }
__device__ __forceinline__ float  bf2f(__bf16 x){ return (float)x; }

// ---------------------------------------------------------------------------
// Kernel 1: routing + per-batch weight combine with BN folding.
// ---------------------------------------------------------------------------
__global__ __launch_bounds__(256)
void k_route_combine(const float* __restrict__ x,  const float* __restrict__ Wr,
                     const float* __restrict__ br, const float* __restrict__ Wpw,
                     const float* __restrict__ bpw,const float* __restrict__ Wdw,
                     const float* __restrict__ bdw,const float* __restrict__ Wpwl,
                     const float* __restrict__ bpwl,
                     const float* __restrict__ g1,const float* __restrict__ b1,
                     const float* __restrict__ m1,const float* __restrict__ v1,
                     const float* __restrict__ g2,const float* __restrict__ b2,
                     const float* __restrict__ m2,const float* __restrict__ v2,
                     const float* __restrict__ g3,const float* __restrict__ b3,
                     const float* __restrict__ m3,const float* __restrict__ v3,
                     __bf16* __restrict__ WkA_t, float* __restrict__ shift1,
                     float* __restrict__ Wdwk,  float* __restrict__ shift2,
                     __bf16* __restrict__ WkB_t, float* __restrict__ shift3)
{
  const int b = blockIdx.x, tid = threadIdx.x;
  __shared__ float part[192];
  __shared__ float pooled[C_];
  __shared__ float rw[E_];
  __shared__ float s1[M_], s2[M_], s3[F_];

  if (tid < 2*C_) {   // global average pool
    const int c = tid % C_, h = tid / C_;
    const float* xp = x + ((size_t)b*P_)*C_ + c;
    float a0=0.f,a1=0.f,a2=0.f,a3=0.f;
    const int p0 = h*(P_/2);
    for (int p = p0; p < p0 + P_/2; p += 4) {
      a0 += xp[(size_t)(p+0)*C_];  a1 += xp[(size_t)(p+1)*C_];
      a2 += xp[(size_t)(p+2)*C_];  a3 += xp[(size_t)(p+3)*C_];
    }
    part[tid] = a0+a1+a2+a3;
  }
  __syncthreads();
  if (tid < C_) pooled[tid] = (part[tid] + part[tid+C_]) * (1.0f/(float)P_);
  __syncthreads();
  if (tid < E_) {
    float s = br[tid];
    for (int c = 0; c < C_; ++c) s += pooled[c]*Wr[c*E_+tid];
    rw[tid] = 1.f/(1.f+__expf(-s));
  }
  for (int m = tid; m < M_; m += 256) {
    s1[m] = g1[m]*rsqrtf(v1[m]+EPS_);
    s2[m] = g2[m]*rsqrtf(v2[m]+EPS_);
  }
  for (int f = tid; f < F_; f += 256) s3[f] = g3[f]*rsqrtf(v3[f]+EPS_);
  __syncthreads();
  const float r0=rw[0], r1=rw[1], r2=rw[2], r3=rw[3];

  for (int i = tid; i < M_*C_; i += 256) {          // WkA_t (transposed, s1-folded)
    const int m = i / C_, c = i % C_;
    const size_t o = (size_t)c*M_ + m, s = (size_t)C_*M_;
    const float w = r0*Wpw[o] + r1*Wpw[o+s] + r2*Wpw[o+2*s] + r3*Wpw[o+3*s];
    WkA_t[((size_t)b*M_ + m)*C_ + c] = f2bf(w * s1[m]);
  }
  for (int m = tid; m < M_; m += 256) {
    const float bb = r0*bpw[m] + r1*bpw[M_+m] + r2*bpw[2*M_+m] + r3*bpw[3*M_+m];
    shift1[(size_t)b*M_+m] = bb*s1[m] + (b1[m]-m1[m]*s1[m]);
  }
  for (int i = tid; i < 9*M_; i += 256) {           // depthwise kernel, s2-folded
    const int m = i % M_;
    const float w = r0*Wdw[i] + r1*Wdw[9*M_+i] + r2*Wdw[18*M_+i] + r3*Wdw[27*M_+i];
    Wdwk[(size_t)b*9*M_ + i] = w * s2[m];
  }
  for (int m = tid; m < M_; m += 256) {
    const float bb = r0*bdw[m] + r1*bdw[M_+m] + r2*bdw[2*M_+m] + r3*bdw[3*M_+m];
    shift2[(size_t)b*M_+m] = bb*s2[m] + (b2[m]-m2[m]*s2[m]);
  }
  for (int i = tid; i < F_*M_; i += 256) {          // WkB_t (transposed, s3-folded)
    const int f = i / M_, m = i % M_;
    const size_t o = (size_t)m*F_ + f, s = (size_t)M_*F_;
    const float w = r0*Wpwl[o] + r1*Wpwl[o+s] + r2*Wpwl[o+2*s] + r3*Wpwl[o+3*s];
    WkB_t[((size_t)b*F_ + f)*M_ + m] = f2bf(w * s3[f]);
  }
  for (int f = tid; f < F_; f += 256) {
    const float bb = r0*bpwl[f] + r1*bpwl[F_+f] + r2*bpwl[2*F_+f] + r3*bpwl[3*F_+f];
    shift3[(size_t)b*F_+f] = bb*s3[f] + (b3[f]-m3[f]*s3[f]);
  }
}

// ---------------------------------------------------------------------------
// Kernel 2: pw1 GEMM  h1[b] = relu( X(3136x96) @ WkA'(96x384) + shift1 ), bf16.
// Full-K preload (K=96): one barrier, then 6 barrier-free WMMAs per wave.
// LDS rows padded to 104 bf16 -> conflict-free b128 fragment reads.
// ---------------------------------------------------------------------------
__global__ __launch_bounds__(256)
void k_pw1(const float* __restrict__ x, const __bf16* __restrict__ WkA_t,
           const float* __restrict__ shift1, __bf16* __restrict__ h1)
{
  constexpr int AS = 104;                 // 96 + 8 pad
  __shared__ __bf16 At[64*AS];            // 13312 B
  __shared__ __bf16 Bt[64*AS];            // 13312 B
  const int nt = blockIdx.x, pt = blockIdx.y, b = blockIdx.z;
  const int tid = threadIdx.x, lane = tid & 31, wid = tid >> 5;
  const int waveM = wid & 3, waveN = wid >> 2;
  const int l15 = lane & 15, lhi = lane >> 4;

  // stage A: 64 pixels x 96 ch, f32 -> bf16 (1536 float4, 6 per thread)
  #pragma unroll
  for (int s = 0; s < 6; ++s) {
    const int f4 = s*256 + tid;
    const int row = f4 / 24, col = (f4 % 24) * 4;
    const float4 u = *(const float4*)(x + (((size_t)b*P_ + pt*64 + row)*C_ + col));
    __bf16* d = &At[row*AS + col];
    d[0]=f2bf(u.x); d[1]=f2bf(u.y); d[2]=f2bf(u.z); d[3]=f2bf(u.w);
  }
  // stage B: 64 n-rows x 96 k (768 x 16B segs, 3 per thread)
  #pragma unroll
  for (int s = 0; s < 3; ++s) {
    const int i = s*256 + tid;
    const int row = i / 12, seg = i % 12;
    const uint4* src = (const uint4*)(WkA_t + (((size_t)b*M_ + nt*64 + row)*C_ + seg*8));
    *(uint4*)&Bt[row*AS + seg*8] = *src;
  }
  __syncthreads();

  v8f acc0 = {}, acc1 = {};
  const int arow = waveM*16 + l15;
  const int c0  = lhi*8;    // A: lanes 16-31 hold K+8
  const int k0b = lhi*8;    // B: lanes 16-31 hold K+8
  const int n0  = nt*64 + waveN*32 + l15;

  #pragma unroll
  for (int ks = 0; ks < 3; ++ks) {
    const v8bf alo = *(const v8bf*)&At[arow*AS + ks*32 + c0];
    const v8bf ahi = *(const v8bf*)&At[arow*AS + ks*32 + 16 + c0];
    v16bf a;
    #pragma unroll
    for (int j=0;j<8;++j){ a[j]=alo[j]; a[j+8]=ahi[j]; }
    #pragma unroll
    for (int sub = 0; sub < 2; ++sub) {
      const int brow = waveN*32 + sub*16 + l15;
      const v8bf blo = *(const v8bf*)&Bt[brow*AS + ks*32 + k0b];
      const v8bf bhi = *(const v8bf*)&Bt[brow*AS + ks*32 + k0b + 16];
      v16bf bb;
      #pragma unroll
      for (int j=0;j<8;++j){ bb[j]=blo[j]; bb[j+8]=bhi[j]; }
      if (sub == 0)
        acc0 = __builtin_amdgcn_wmma_f32_16x16x32_bf16(false,a,false,bb,(short)0,acc0,false,false);
      else
        acc1 = __builtin_amdgcn_wmma_f32_16x16x32_bf16(false,a,false,bb,(short)0,acc1,false,false);
    }
  }
  const float sh0 = shift1[(size_t)b*M_ + n0];
  const float sh1 = shift1[(size_t)b*M_ + n0 + 16];
  #pragma unroll
  for (int r = 0; r < 8; ++r) {
    const int mrow = waveM*16 + r + lhi*8;   // C/D: lanes 16-31 are rows+8
    const size_t base = ((size_t)b*P_ + pt*64 + mrow)*M_;
    h1[base + n0]      = f2bf(fmaxf(acc0[r] + sh0, 0.f));
    h1[base + n0 + 16] = f2bf(fmaxf(acc1[r] + sh1, 0.f));
  }
}

// ---------------------------------------------------------------------------
// Kernel 3: fused depthwise3x3(+BN2,relu) -> LDS tile -> pw2 GEMM (+BN3,+res).
// Whole per-batch B (96x384 bf16) preloaded to LDS -> GEMM loop is barrier-free.
// LDS: h2 tile 50.2KB + B 75.3KB + dw weights 13.8KB + shifts 1.5KB = 140.8KB
// (2 workgroups per 320KB WGP).
// ---------------------------------------------------------------------------
__global__ __launch_bounds__(256)
void k_dw_pw2(const __bf16* __restrict__ h1, const float* __restrict__ Wdwk,
              const float* __restrict__ shift2, const __bf16* __restrict__ WkB_t,
              const float* __restrict__ shift3, const float* __restrict__ x,
              float* __restrict__ out)
{
  constexpr int HS = 392;   // padded stride (conflict-free b128 reads)
  __shared__ __bf16 h2t[64*HS];     // 50176 B
  __shared__ __bf16 Bfull[96*HS];   // 75264 B
  __shared__ float  kw[9*M_];       // 13824 B
  __shared__ float  sh2[M_];        //  1536 B

  const int tx = blockIdx.x, ty = blockIdx.y, b = blockIdx.z;
  const int tid = threadIdx.x, lane = tid & 31, wid = tid >> 5;
  const int waveM = wid & 3, waveN = wid >> 2;
  const int l15 = lane & 15, lhi = lane >> 4;

  // stage full B: 96 rows x 384 bf16 = 4608 x 16B segs, 18 per thread.
  // Issued first; all of it lands in LDS before the barrier ahead of the depthwise stage.
  #pragma unroll
  for (int s = 0; s < 18; ++s) {
    const int i = s*256 + tid;
    const int row = i / 48, seg = i % 48;
    const uint4* src = (const uint4*)(WkB_t + (((size_t)b*F_ + row)*M_ + seg*8));
    *(uint4*)&Bfull[row*HS + seg*8] = *src;
  }
  for (int i = tid; i < 9*M_; i += 256) kw[i] = Wdwk[(size_t)b*9*M_ + i];
  for (int i = tid; i < M_;   i += 256) sh2[i] = shift2[(size_t)b*M_ + i];
  __syncthreads();

  // depthwise 3x3 (s2-folded) + shift2 + relu -> h2 tile in LDS (bf16)
  for (int i = tid; i < 64*M_; i += 256) {
    const int pix = i / M_, m = i % M_;
    const int gy0 = ty*8 + (pix >> 3), gx0 = tx*8 + (pix & 7);
    float a = 0.f;
    #pragma unroll
    for (int tap = 0; tap < 9; ++tap) {
      const int gy = gy0 + tap/3 - 1, gx = gx0 + tap%3 - 1;
      if (gy >= 0 && gy < HH_ && gx >= 0 && gx < WW_)
        a += bf2f(h1[((size_t)b*P_ + gy*WW_ + gx)*M_ + m]) * kw[tap*M_ + m];
    }
    h2t[pix*HS + m] = f2bf(fmaxf(a + sh2[m], 0.f));
  }
  __syncthreads();

  // barrier-free GEMM: 12 K-steps, 3 N-frags per wave, 36 WMMAs
  v8f acc[3] = {};
  const int arow = waveM*16 + l15;
  const int c0 = lhi*8, k0b = lhi*8;
  for (int ks = 0; ks < 12; ++ks) {
    const v8bf alo = *(const v8bf*)&h2t[arow*HS + ks*32 + c0];
    const v8bf ahi = *(const v8bf*)&h2t[arow*HS + ks*32 + 16 + c0];
    v16bf a;
    #pragma unroll
    for (int j=0;j<8;++j){ a[j]=alo[j]; a[j+8]=ahi[j]; }
    #pragma unroll
    for (int sub = 0; sub < 3; ++sub) {
      const int nrow = waveN*48 + sub*16 + l15;
      const v8bf blo = *(const v8bf*)&Bfull[nrow*HS + ks*32 + k0b];
      const v8bf bhi = *(const v8bf*)&Bfull[nrow*HS + ks*32 + k0b + 16];
      v16bf bb;
      #pragma unroll
      for (int j=0;j<8;++j){ bb[j]=blo[j]; bb[j+8]=bhi[j]; }
      acc[sub] = __builtin_amdgcn_wmma_f32_16x16x32_bf16(false,a,false,bb,(short)0,acc[sub],false,false);
    }
  }

  #pragma unroll
  for (int sub = 0; sub < 3; ++sub) {
    const int n = waveN*48 + sub*16 + l15;
    const float sh = shift3[(size_t)b*F_ + n];
    #pragma unroll
    for (int r = 0; r < 8; ++r) {
      const int pix = waveM*16 + r + lhi*8;
      const int gp = (ty*8 + (pix >> 3))*WW_ + tx*8 + (pix & 7);
      const size_t o = ((size_t)b*P_ + gp)*F_ + n;
      out[o] = acc[sub][r] + sh + x[o];
    }
  }
}

// ---------------------------------------------------------------------------
extern "C" void kernel_launch(void* const* d_in, const int* in_sizes, int n_in,
                              void* d_out, int out_size, void* d_ws, size_t ws_size,
                              hipStream_t stream) {
  const float* x    = (const float*)d_in[0];
  const float* Wr   = (const float*)d_in[1];
  const float* br   = (const float*)d_in[2];
  const float* Wpw  = (const float*)d_in[3];
  const float* bpw  = (const float*)d_in[4];
  const float* Wdw  = (const float*)d_in[5];
  const float* bdw  = (const float*)d_in[6];
  const float* Wpwl = (const float*)d_in[7];
  const float* bpwl = (const float*)d_in[8];
  const float* g1 = (const float*)d_in[9];  const float* b1 = (const float*)d_in[10];
  const float* m1 = (const float*)d_in[11]; const float* v1 = (const float*)d_in[12];
  const float* g2 = (const float*)d_in[13]; const float* b2 = (const float*)d_in[14];
  const float* m2 = (const float*)d_in[15]; const float* v2 = (const float*)d_in[16];
  const float* g3 = (const float*)d_in[17]; const float* b3 = (const float*)d_in[18];
  const float* m3 = (const float*)d_in[19]; const float* v3 = (const float*)d_in[20];

  char* ws = (char*)d_ws;
  __bf16* WkA_t  = (__bf16*)(ws + 0);                 // 2359296
  float*  shift1 = (float*)(ws + 2359296);            //   49152
  float*  Wdwk   = (float*)(ws + 2408448);            //  442368
  float*  shift2 = (float*)(ws + 2850816);            //   49152
  __bf16* WkB_t  = (__bf16*)(ws + 2899968);           // 2359296
  float*  shift3 = (float*)(ws + 5259264);            //   12288
  __bf16* h1     = (__bf16*)(ws + 5271552);           // 77070336
  // total: 82341888 bytes

  k_route_combine<<<B_, 256, 0, stream>>>(
      x, Wr, br, Wpw, bpw, Wdw, bdw, Wpwl, bpwl,
      g1,b1,m1,v1, g2,b2,m2,v2, g3,b3,m3,v3,
      WkA_t, shift1, Wdwk, shift2, WkB_t, shift3);

  k_pw1<<<dim3(M_/64, P_/64, B_), 256, 0, stream>>>(x, WkA_t, shift1, h1);

  k_dw_pw2<<<dim3(WW_/8, HH_/8, B_), 256, 0, stream>>>(
      h1, Wdwk, shift2, WkB_t, shift3, x, (float*)d_out);
}
